// LlamaAttentionPNA_19164144074842
// MI455X (gfx1250) — matrix operand order repaired and checked by reading, not compiled, not run on hardware
//
#include <hip/hip_runtime.h>
#include <hip/hip_bf16.h>

typedef __attribute__((ext_vector_type(16))) _Float16 v16h;
typedef __attribute__((ext_vector_type(8)))  _Float16 v8h;
typedef __attribute__((ext_vector_type(4)))  _Float16 v4h;
typedef __attribute__((ext_vector_type(8)))  float    v8f;

#define SEQ  1024
#define NH   16
#define DH   64
#define HID_ 1024
#define DM_  128
#define F4D  256   // 4*DH

// ---------------------------------------------------------------------------
// f32 -> f16 elementwise convert
// ---------------------------------------------------------------------------
__global__ __launch_bounds__(256) void cvt_f32_f16_kernel(const float* __restrict__ src,
                                                          _Float16* __restrict__ dst, int n) {
  int i = blockIdx.x * 256 + threadIdx.x;
  if (i < n) dst[i] = (_Float16)src[i];
}

// ---------------------------------------------------------------------------
// RoPE + reshape (B,S,H,D) f32 -> (B,H,S,D) f16
// ---------------------------------------------------------------------------
__global__ __launch_bounds__(256) void rope_reshape_kernel(const float* __restrict__ src,
                                                           _Float16* __restrict__ dst) {
  int idx = blockIdx.x * 256 + threadIdx.x;   // exact grid: B*S*H*D
  int d  = idx & 63;
  int hh = (idx >> 6) & 15;
  int s  = (idx >> 10) & 1023;
  int b  = idx >> 20;
  float val = src[idx];
  float par = src[idx + ((d < 32) ? 32 : -32)];
  float rot = (d < 32) ? -par : par;
  int   fi  = d & 31;
  float inv = __powf(10000.0f, -(2.0f * (float)fi) / 64.0f);
  float ang = (float)s * inv;
  float o = val * __cosf(ang) + rot * __sinf(ang);
  dst[(((size_t)(b * NH + hh)) * SEQ + s) * DH + d] = (_Float16)o;
}

// reshape only, f32 out (for v)
__global__ __launch_bounds__(256) void reshape_v_kernel(const float* __restrict__ src,
                                                        float* __restrict__ dst) {
  int idx = blockIdx.x * 256 + threadIdx.x;
  int d  = idx & 63;
  int hh = (idx >> 6) & 15;
  int s  = (idx >> 10) & 1023;
  int b  = idx >> 20;
  dst[(((size_t)(b * NH + hh)) * SEQ + s) * DH + d] = src[idx];
}

// ---------------------------------------------------------------------------
// WMMA GEMM: C(MxN,f32) = A(MxK,f16 rowmajor) * W(NxK,f16 rowmajor)^T
// 128x64 tile per block, 256 threads = 8 waves (4x2 wave grid).
// Each wave owns a 32x32 sub-tile = 2x2 accumulators; K-step 64
// -> 8 v_wmma per wave per barrier pair.
// ---------------------------------------------------------------------------
__global__ __launch_bounds__(256) void gemm_abT_kernel(const _Float16* __restrict__ A,
                                                       const _Float16* __restrict__ W,
                                                       float* __restrict__ C,
                                                       int M, int N, int K) {
  __shared__ _Float16 la[128 * 72];   // 128 rows x 64 (K) halves, pad 8
  __shared__ _Float16 lb[64 * 72];    //  64 rows x 64 (K) halves, pad 8
  const int tid  = threadIdx.x;
  const int lane = tid & 31;
  const int wave = tid >> 5;
  const int m0 = blockIdx.x * 128;
  const int n0 = blockIdx.y * 64;
  const int wr = wave >> 1;           // 0..3 -> 32-row band
  const int wc = wave & 1;            // 0..1 -> 32-col band
  const int nl  = lane & 15;
  const int kbA = (lane < 16) ? 0 : 16;
  const int kbB = (lane < 16) ? 0 : 16;
  v8f acc[2][2] = {};
  for (int k0 = 0; k0 < K; k0 += 64) {
    // stage A tile: 128x64 halves, 4 x v8h per thread
#pragma unroll
    for (int i = 0; i < 4; ++i) {
      int id = tid + 256 * i;
      int r = id >> 3, cc = (id & 7) * 8;
      *(v8h*)(la + r * 72 + cc) = *(const v8h*)(A + (size_t)(m0 + r) * K + k0 + cc);
    }
    // stage B tile: 64x64 halves, 2 x v8h per thread
#pragma unroll
    for (int i = 0; i < 2; ++i) {
      int id = tid + 256 * i;
      int r = id >> 3, cc = (id & 7) * 8;
      *(v8h*)(lb + r * 72 + cc) = *(const v8h*)(W + (size_t)(n0 + r) * K + k0 + cc);
    }
    if (k0 + 64 < K) {
      __builtin_prefetch(A + (size_t)(m0 + (tid >> 1)) * K + k0 + 64 + (tid & 1) * 32);
      __builtin_prefetch(W + (size_t)(n0 + (tid >> 2)) * K + k0 + 64 + (tid & 3) * 16);
    }
    __syncthreads();
#pragma unroll
    for (int kc = 0; kc < 2; ++kc) {
      const int kof = kc * 32;
      v16h af[2], bf[2];
#pragma unroll
      for (int ms = 0; ms < 2; ++ms) {
#pragma unroll
        for (int e = 0; e < 16; ++e) {
          int k = kof + kbA + e;
          af[ms][e] = la[(wr * 32 + ms * 16 + nl) * 72 + k];
        }
      }
#pragma unroll
      for (int ns = 0; ns < 2; ++ns) {
#pragma unroll
        for (int e = 0; e < 16; ++e) {
          bf[ns][e] = lb[(wc * 32 + ns * 16 + nl) * 72 + kof + kbB + e];
        }
      }
#pragma unroll
      for (int ms = 0; ms < 2; ++ms)
#pragma unroll
        for (int ns = 0; ns < 2; ++ns)
          acc[ms][ns] = __builtin_amdgcn_wmma_f32_16x16x32_f16(
              false, af[ms], false, bf[ns], (short)0, acc[ms][ns], false, false);
    }
    __syncthreads();
  }
#pragma unroll
  for (int ms = 0; ms < 2; ++ms) {
#pragma unroll
    for (int ns = 0; ns < 2; ++ns) {
#pragma unroll
      for (int r = 0; r < 8; ++r) {
        int row = m0 + wr * 32 + ms * 16 + r + ((lane < 16) ? 0 : 8);
        int col = n0 + wc * 32 + ns * 16 + nl;
        C[(size_t)row * N + col] = acc[ms][ns][r];
      }
    }
  }
}

// ---------------------------------------------------------------------------
// PNA kernel: one block per (b, h, 16-row block).
//  - WMMA q.k^T -> thresholded adj row-block in LDS (f16, -1 = ineligible)
//  - exact scaled top-k per row (15-bit binary search on half bits + stable
//    tie-break by ascending index)
//  - sum/mean/max/var aggregation of v over selected neighbors
//  - per-head MLP (WMMA) with SiLU, + eps*v residual -> oh (B,S,H*D) f16
// ---------------------------------------------------------------------------
__global__ __launch_bounds__(256) void pna_kernel(const _Float16* __restrict__ qh,
                                                  const _Float16* __restrict__ kh,
                                                  const float*    __restrict__ vf,
                                                  const _Float16* __restrict__ w1h,
                                                  const _Float16* __restrict__ w2h,
                                                  const float*    __restrict__ res_eps_p,
                                                  _Float16* __restrict__ oh) {
  __shared__ _Float16 adjb[16 * SEQ];     // 32 KB
  __shared__ _Float16 qt[16 * 72];
  __shared__ unsigned selw[16 * 32];
  __shared__ _Float16 hc[16 * 264];       // hcat 16 x 256 (padded)
  __shared__ _Float16 mid[16 * 136];      // 16 x 128 (padded)

  const int tid  = threadIdx.x;
  const int lane = tid & 31;
  const int wave = tid >> 5;
  const int blk = blockIdx.x;
  const int rb = blk & 63;
  const int hh = (blk >> 6) & 15;
  const int b  = blk >> 10;
  const int s0 = rb * 16;
  const size_t hb = ((size_t)(b * NH + hh)) * SEQ * DH;

  for (int i = tid; i < 16 * SEQ; i += 256) adjb[i] = (_Float16)(-1.0f);
  {
    int r = tid >> 4, c4 = (tid & 15) * 4;
    *(v4h*)(qt + r * 72 + c4) = *(const v4h*)(qh + hb + (size_t)(s0 + r) * DH + c4);
  }
  __syncthreads();

  const int nl  = lane & 15;
  const int kbA = (lane < 16) ? 0 : 16;
  const int kbB = (lane < 16) ? 0 : 16;

  // ---- adj = relu(q.k^T / 8), causal-strict, thresholded ----
  v16h aq0, aq1;
#pragma unroll
  for (int e = 0; e < 16; ++e) {
    int k = kbA + e;
    aq0[e] = qt[nl * 72 + k];
    aq1[e] = qt[nl * 72 + 32 + k];
  }
  const int ntiles = rb + 1;
  for (int jt = wave; jt < ntiles; jt += 8) {
    const int j0 = jt * 16;
    v8f c = {};
    {
      const _Float16* kp = kh + hb + (size_t)(j0 + nl) * DH + kbB;
      v16h bf;
      v8h p0 = *(const v8h*)kp, p1 = *(const v8h*)(kp + 8);
#pragma unroll
      for (int e = 0; e < 8; ++e) { bf[e] = p0[e]; bf[8 + e] = p1[e]; }
      c = __builtin_amdgcn_wmma_f32_16x16x32_f16(false, aq0, false, bf, (short)0, c, false, false);
    }
    {
      const _Float16* kp = kh + hb + (size_t)(j0 + nl) * DH + 32 + kbB;
      v16h bf;
      v8h p0 = *(const v8h*)kp, p1 = *(const v8h*)(kp + 8);
#pragma unroll
      for (int e = 0; e < 8; ++e) { bf[e] = p0[e]; bf[8 + e] = p1[e]; }
      c = __builtin_amdgcn_wmma_f32_16x16x32_f16(false, aq1, false, bf, (short)0, c, false, false);
    }
#pragma unroll
    for (int r = 0; r < 8; ++r) {
      int lr = r + ((lane < 16) ? 0 : 8);
      int sq = s0 + lr;
      int j  = j0 + nl;
      float x = c[r] * 0.125f;
      x = (x > 0.0f) ? x : 0.0f;
      float f = (j < sq) ? ((x >= 0.2f) ? x : 0.0f) : -1.0f;
      adjb[lr * SEQ + j] = (_Float16)f;
    }
  }
  __syncthreads();

  // ---- per-row top-k selection + aggregation (2 rows per wave) ----
  for (int rr = 0; rr < 2; ++rr) {
    const int lr = wave * 2 + rr;
    const int sq = s0 + lr;
    int kk = 0;
    if (sq > 0) {
      kk = (int)ceilf(0.1f * (float)sq);
      if (kk < 1) kk = 1;
    }
    unsigned pref = 0;
    if (kk > 0) {
      for (int bit = 14; bit >= 0; --bit) {
        unsigned cand = pref | (1u << bit);
        float cf = (float)__builtin_bit_cast(_Float16, (unsigned short)cand);
        int cnt = 0;
        for (int ch = 0; ch < 32; ++ch) {
          float x = (float)adjb[lr * SEQ + ch * 32 + lane];
          cnt += __popc((unsigned)__ballot(x >= cf));
        }
        if (cnt >= kk) pref = cand;
      }
    }
    const float vk = (float)__builtin_bit_cast(_Float16, (unsigned short)pref);
    int ngt = 0;
    for (int ch = 0; ch < 32; ++ch) {
      float x = (float)adjb[lr * SEQ + ch * 32 + lane];
      ngt += __popc((unsigned)__ballot((kk > 0) && (x > vk)));
    }
    const int needeq = kk - ngt;
    int eqrun = 0;
    for (int ch = 0; ch < 32; ++ch) {
      float x = (float)adjb[lr * SEQ + ch * 32 + lane];
      bool isgt = (kk > 0) && (x > vk);
      bool iseq = (kk > 0) && (x == vk);
      unsigned beq = (unsigned)__ballot(iseq);
      unsigned premask = (lane == 0) ? 0u : ((1u << lane) - 1u);
      bool sel = isgt || (iseq && (eqrun + __popc(beq & premask) < needeq));
      unsigned w = (unsigned)__ballot(sel);
      if (lane == 0) selw[lr * 32 + ch] = w;
      eqrun += __popc(beq);
    }
    // aggregation: each lane owns dims (2*lane, 2*lane+1)
    const int d0 = lane * 2;
    const float* vrow = vf + hb;
    float s0a = 0.f, s1a = 0.f, q0a = 0.f, q1a = 0.f, m0a = -3.0e38f, m1a = -3.0e38f;
    for (int ch = 0; ch < 32; ++ch) {
      unsigned w = selw[lr * 32 + ch];
      while (w) {
        int bit = __ffs(w) - 1;
        w &= (w - 1);
        int j = ch * 32 + bit;
        float2 vv = *(const float2*)(vrow + (size_t)j * DH + d0);
        s0a += vv.x; s1a += vv.y;
        q0a += vv.x * vv.x; q1a += vv.y * vv.y;
        m0a = fmaxf(m0a, vv.x); m1a = fmaxf(m1a, vv.y);
      }
    }
    float deg = (float)kk;
    float dsf = (deg > 0.f) ? deg : 1.f;
    float mean0 = s0a / dsf, mean1 = s1a / dsf;
    float var0 = fmaxf(q0a / dsf - mean0 * mean0, 0.f);
    float var1 = fmaxf(q1a / dsf - mean1 * mean1, 0.f);
    if (deg <= 0.f) { m0a = 0.f; m1a = 0.f; }
    _Float16* hr = hc + lr * 264;
    hr[d0] = (_Float16)s0a;            hr[d0 + 1] = (_Float16)s1a;
    hr[64 + d0] = (_Float16)mean0;     hr[64 + d0 + 1] = (_Float16)mean1;
    hr[128 + d0] = (_Float16)m0a;      hr[128 + d0 + 1] = (_Float16)m1a;
    hr[192 + d0] = (_Float16)var0;     hr[192 + d0 + 1] = (_Float16)var1;
  }
  __syncthreads();

  // ---- MLP layer 1: mid = silu(hcat @ W1[h]) -- 16x256 * 256x128 ----
  {
    v8f c = {};
#pragma unroll
    for (int ks = 0; ks < 8; ++ks) {
      v16h af, bf;
#pragma unroll
      for (int e = 0; e < 16; ++e) {
        int k = ks * 32 + kbA + e;
        af[e] = hc[nl * 264 + k];
      }
#pragma unroll
      for (int e = 0; e < 16; ++e) {
        int k = ks * 32 + kbB + e;
        bf[e] = w1h[(size_t)hh * F4D * DM_ + (size_t)k * DM_ + wave * 16 + nl];
      }
      c = __builtin_amdgcn_wmma_f32_16x16x32_f16(false, af, false, bf, (short)0, c, false, false);
    }
#pragma unroll
    for (int r = 0; r < 8; ++r) {
      int lr = r + ((lane < 16) ? 0 : 8);
      float x = c[r];
      float sl = x / (1.0f + __expf(-x));
      mid[lr * 136 + wave * 16 + nl] = (_Float16)sl;
    }
  }
  __syncthreads();

  // ---- MLP layer 2: out = mid @ W2[h] + eps*v  -- 16x128 * 128x64 ----
  if (wave < 4) {
    v8f c = {};
#pragma unroll
    for (int ks = 0; ks < 4; ++ks) {
      v16h af, bf;
#pragma unroll
      for (int e = 0; e < 16; ++e) {
        int k = ks * 32 + kbA + e;
        af[e] = mid[nl * 136 + k];
      }
#pragma unroll
      for (int e = 0; e < 16; ++e) {
        int k = ks * 32 + kbB + e;
        bf[e] = w2h[(size_t)hh * DM_ * DH + (size_t)k * DH + wave * 16 + nl];
      }
      c = __builtin_amdgcn_wmma_f32_16x16x32_f16(false, af, false, bf, (short)0, c, false, false);
    }
    const float eps = res_eps_p[0];
#pragma unroll
    for (int r = 0; r < 8; ++r) {
      int lr = r + ((lane < 16) ? 0 : 8);
      int sq = s0 + lr;
      int d  = wave * 16 + nl;
      float val = c[r] + eps * vf[hb + (size_t)sq * DH + d];
      oh[((size_t)(b * SEQ + sq)) * HID_ + hh * 64 + d] = (_Float16)val;
    }
  }
}

// ---------------------------------------------------------------------------
// Host-side launch
// ---------------------------------------------------------------------------
extern "C" void kernel_launch(void* const* d_in, const int* in_sizes, int n_in,
                              void* d_out, int out_size, void* d_ws, size_t ws_size,
                              hipStream_t stream) {
  (void)in_sizes; (void)n_in; (void)out_size;
  const float* x   = (const float*)d_in[0];
  const float* Wq  = (const float*)d_in[1];
  const float* Wk  = (const float*)d_in[2];
  const float* Wv  = (const float*)d_in[3];
  const float* Wo  = (const float*)d_in[4];
  const float* W1  = (const float*)d_in[5];
  const float* W2  = (const float*)d_in[6];
  const float* eps = (const float*)d_in[7];
  float* out = (float*)d_out;
  char* ws = (char*)d_ws;

  const int M = 2 * SEQ;           // 2048
  const int NELT = 2 * SEQ * HID_; // 2,097,152

  size_t off = 0;
  auto take = [&](size_t bytes) { size_t o = off; off += (bytes + 255) & ~(size_t)255; return o; };
  size_t o_xh  = take((size_t)NELT * 2);
  size_t o_wq  = take((size_t)HID_ * HID_ * 2);
  size_t o_wk  = take((size_t)HID_ * HID_ * 2);
  size_t o_wv  = take((size_t)HID_ * HID_ * 2);
  size_t o_wo  = take((size_t)HID_ * HID_ * 2);
  size_t o_w1  = take((size_t)NH * F4D * DM_ * 2);
  size_t o_w2  = take((size_t)NH * DM_ * DH * 2);
  size_t o_qf  = take((size_t)NELT * 4);
  size_t o_kf  = take((size_t)NELT * 4);
  size_t o_vr  = take((size_t)NELT * 4);
  size_t o_qh  = take((size_t)NELT * 2);
  size_t o_kh  = take((size_t)NELT * 2);
  size_t o_vf  = take((size_t)NELT * 4);
  size_t o_oh  = take((size_t)NELT * 2);
  if (off > ws_size) return;

  _Float16* xh  = (_Float16*)(ws + o_xh);
  _Float16* wqh = (_Float16*)(ws + o_wq);
  _Float16* wkh = (_Float16*)(ws + o_wk);
  _Float16* wvh = (_Float16*)(ws + o_wv);
  _Float16* woh = (_Float16*)(ws + o_wo);
  _Float16* w1h = (_Float16*)(ws + o_w1);
  _Float16* w2h = (_Float16*)(ws + o_w2);
  float*    qf  = (float*)(ws + o_qf);
  float*    kf  = (float*)(ws + o_kf);
  float*    vr  = (float*)(ws + o_vr);
  _Float16* qh  = (_Float16*)(ws + o_qh);
  _Float16* kh  = (_Float16*)(ws + o_kh);
  float*    vf  = (float*)(ws + o_vf);
  _Float16* oh  = (_Float16*)(ws + o_oh);

  auto cvt = [&](const float* s, _Float16* d, int n) {
    cvt_f32_f16_kernel<<<(n + 255) / 256, 256, 0, stream>>>(s, d, n);
  };
  cvt(x,  xh,  NELT);
  cvt(Wq, wqh, HID_ * HID_);
  cvt(Wk, wkh, HID_ * HID_);
  cvt(Wv, wvh, HID_ * HID_);
  cvt(Wo, woh, HID_ * HID_);
  cvt(W1, w1h, NH * F4D * DM_);
  cvt(W2, w2h, NH * DM_ * DH);

  dim3 gg(M / 128, HID_ / 64);
  gemm_abT_kernel<<<gg, 256, 0, stream>>>(xh, wqh, qf, M, HID_, HID_);
  gemm_abT_kernel<<<gg, 256, 0, stream>>>(xh, wkh, kf, M, HID_, HID_);
  gemm_abT_kernel<<<gg, 256, 0, stream>>>(xh, wvh, vr, M, HID_, HID_);

  rope_reshape_kernel<<<NELT / 256, 256, 0, stream>>>(qf, qh);
  rope_reshape_kernel<<<NELT / 256, 256, 0, stream>>>(kf, kh);
  reshape_v_kernel  <<<NELT / 256, 256, 0, stream>>>(vr, vf);

  pna_kernel<<<2 * NH * (SEQ / 16), 256, 0, stream>>>(qh, kh, vf, w1h, w2h, eps, oh);

  gemm_abT_kernel<<<gg, 256, 0, stream>>>(oh, woh, out, M, HID_, HID_);
}
